// AttentionBlock_14164802143008
// MI455X (gfx1250) — matrix operand order repaired and checked by reading, not compiled, not run on hardware
//
#include <hip/hip_runtime.h>

// ---------------------------------------------------------------------------
// Fused attention block for MI455X (gfx1250, wave32, WMMA 16x16x32 f16).
//   proj_qkv    : per-voxel channel GEMM (1x1x1 conv) -> f16 q/k/v workspace
//   flash_attn  : fused softmax(Q^T K) V, K/V tiles streamed by the Tensor
//                 Data Mover into double-buffered LDS (TENSORcnt pipelined,
//                 one quarter-tile DMA per wave -> no EXEC-masked TDM issue)
// ---------------------------------------------------------------------------

typedef __attribute__((ext_vector_type(16))) _Float16 v16h;
typedef __attribute__((ext_vector_type(8)))  _Float16 v8h;
typedef __attribute__((ext_vector_type(8)))  float    v8f;
typedef __attribute__((ext_vector_type(4)))  unsigned int v4u;
typedef __attribute__((ext_vector_type(8)))  int      v8i;
typedef __attribute__((ext_vector_type(4)))  int      v4i;

#define BATCH 2
#define CCH   64      // channels == head dim
#define NTOK  4096    // 16*16*16 voxels
#define BR    64      // query rows per workgroup (16 per wave)
#define BC    32      // key/value columns per inner step
#define FLASH_WAVES 4

union HV {
  v16h v;
  v8h  h[2];
};

// ---------------------------------------------------------------------------
// Projection: q = Wq x + bq (and k, v).  x: [B][C][N] f32.
// Writes: qT,kT as f16 [B][N][C]; v as f16 [B][C][N].
// ---------------------------------------------------------------------------
__global__ __launch_bounds__(256) void proj_qkv_kernel(
    const float* __restrict__ x,
    const float* __restrict__ Wq, const float* __restrict__ bq,
    const float* __restrict__ Wk, const float* __restrict__ bk,
    const float* __restrict__ Wv, const float* __restrict__ bv,
    _Float16* __restrict__ qT, _Float16* __restrict__ kT,
    _Float16* __restrict__ vC)
{
  __shared__ float xs[CCH * 64];          // x tile [c][ii]
  __shared__ float wsh[3 * CCH * CCH];    // Wq | Wk | Wv

  const int t  = threadIdx.x;
  const int b  = blockIdx.y;
  const int i0 = blockIdx.x * 64;

  #pragma unroll
  for (int rep = 0; rep < 16; ++rep) {
    int idx = t + 256 * rep;              // 0..4095
    int c   = idx >> 6;
    int ii  = idx & 63;
    xs[idx]         = x[(((size_t)b * CCH + c) << 12) + i0 + ii];
    wsh[idx]        = Wq[idx];
    wsh[4096 + idx] = Wk[idx];
    wsh[8192 + idx] = Wv[idx];
  }
  __syncthreads();

  const int ii = t & 63;                  // voxel within tile
  const int o0 = (t >> 6) << 4;           // 16 output channels per thread

  float accq[16], acck[16], accv[16];
  #pragma unroll
  for (int r = 0; r < 16; ++r) { accq[r] = 0.f; acck[r] = 0.f; accv[r] = 0.f; }

  for (int c = 0; c < CCH; ++c) {
    float xv = xs[c * 64 + ii];
    #pragma unroll
    for (int r = 0; r < 16; ++r) {
      accq[r] = fmaf(wsh[(o0 + r) * 64 + c],        xv, accq[r]);
      acck[r] = fmaf(wsh[4096 + (o0 + r) * 64 + c], xv, acck[r]);
      accv[r] = fmaf(wsh[8192 + (o0 + r) * 64 + c], xv, accv[r]);
    }
  }

  size_t rowQ = (((size_t)b * NTOK) + i0 + ii) * CCH + o0;
  #pragma unroll
  for (int g = 0; g < 2; ++g) {
    v8h pq, pk;
    #pragma unroll
    for (int e = 0; e < 8; ++e) {
      int r = g * 8 + e;
      pq[e] = (_Float16)(accq[r] + bq[o0 + r]);
      pk[e] = (_Float16)(acck[r] + bk[o0 + r]);
    }
    *(v8h*)(qT + rowQ + 8 * g) = pq;
    *(v8h*)(kT + rowQ + 8 * g) = pk;
  }
  #pragma unroll
  for (int r = 0; r < 16; ++r)
    vC[(((size_t)b * CCH) + o0 + r) * (size_t)NTOK + i0 + ii] =
        (_Float16)(accv[r] + bv[o0 + r]);
}

// ---------------------------------------------------------------------------
// TDM: issue a 2D tile load (data_size = 2 bytes) into LDS.
// D# per CDNA5 ISA 8.3/8.4: group0 = {flags, lds_addr, global_addr, type=2},
// group1 = {mask/data_size, tensor dims, tile dims, dim0 stride}; groups 2/3
// zero (2D tile: tile_dim2 = 0).  6-arg builtin form (clang-23 toolchain).
// ---------------------------------------------------------------------------
__device__ __forceinline__ void tdm_load_2d_f16(
    const _Float16* gptr, unsigned ldsAddr,
    unsigned tileX, unsigned tileY,
    unsigned dimX, unsigned dimY, unsigned long long strideX)
{
  unsigned long long ga = (unsigned long long)gptr;
  v4u g0;
  g0[0] = 1u;                                          // count=1 (valid)
  g0[1] = ldsAddr;                                     // lds_addr [63:32]
  g0[2] = (unsigned)ga;                                // global_addr lo
  g0[3] = (unsigned)((ga >> 32) & 0x01ffffffu)         // global_addr hi
        | (2u << 30);                                  // type=2 [127:126]
  v8i g1;
  g1[0] = (int)(1u << 16);                             // data_size=1 -> 2B
  g1[1] = (int)((dimX & 0xffffu) << 16);               // tensor_dim0 lo16
  g1[2] = (int)(((dimX >> 16) & 0xffffu) | ((dimY & 0xffffu) << 16));
  g1[3] = (int)(((dimY >> 16) & 0xffffu) | (tileX << 16));   // tile_dim0
  g1[4] = (int)(tileY & 0xffffu);                      // tile_dim1 (dim2=0)
  g1[5] = (int)(strideX & 0xffffffffu);                // dim0 stride lo32
  g1[6] = (int)((strideX >> 32) & 0xffffu);            // dim0 stride hi16
  g1[7] = 0;
  v4i gz4 = {0, 0, 0, 0};
  v8i gz8 = {0, 0, 0, 0, 0, 0, 0, 0};
  __builtin_amdgcn_tensor_load_to_lds(g0, g1, gz4, gz4, gz8, 0);
}

// ---------------------------------------------------------------------------
// Fused flash attention.
//   S[i][j] = sum_c q[c,i] k[c,j]  -> A = qT rows (i x c), B = K tile (c x j)
//   O[i][c] = sum_j P[i][j] v[c,j] -> A = P (i x j),       B = V tile (j x c)
// Each of the 4 waves DMAs one quarter of the K and V tiles (TDM ignores
// EXEC, so per-wave issue avoids duplicate loads without scalar branching);
// each wave waits its own TENSORcnt, then the workgroup barrier publishes.
// WMMA operand layouts per CDNA5 ISA 7.12.2 (wave32):
//   A(16x32 f16): lane = m + 16*s ; slots = K {8s..8s+7, 16+8s..16+8s+7}
//   B(32x16 f16): N = lane%16; this code loads slots = K 16h..16h+15 (h = lane/16), and its A loads the same way
//   D(16x16 f32): reg r: M = r + 8*(lane>=16), N = lane%16
// ---------------------------------------------------------------------------
__global__ __launch_bounds__(FLASH_WAVES * 32) void flash_attn_kernel(
    const _Float16* __restrict__ qT,   // [B][N][C]
    const _Float16* __restrict__ kT,   // [B][N][C]
    const _Float16* __restrict__ vC,   // [B][C][N]
    float* __restrict__ out)           // [B][C][N]
{
  __shared__ _Float16 ktile[2][BC][CCH];          // [buf][j][c]   4KB each
  __shared__ _Float16 vtile[2][CCH][BC];          // [buf][c][j]   4KB each
  __shared__ _Float16 pbuf[FLASH_WAVES][16][BC];  // P transpose buffer

  const int lane = threadIdx.x & 31;
  const int wave = threadIdx.x >> 5;
  const int m16  = lane & 15;
  const int hi   = lane >> 4;
  const int b    = blockIdx.y;
  const int i0   = blockIdx.x * BR + wave * 16;

  const _Float16* qBase = qT + ((size_t)b * NTOK) * CCH;
  const _Float16* kBase = kT + ((size_t)b * NTOK) * CCH;
  const _Float16* vBase = vC + ((size_t)b * CCH) * NTOK;

  // Per-wave quarter-tile TDM parameters.
  //   K quarter: rows j = wave*8  .. +8   (8 x 64 elements)
  //   V quarter: rows c = wave*16 .. +16  (16 x 32 elements)
  const size_t   kGOff = (size_t)(wave * 8) * CCH;
  const size_t   vGOff = (size_t)(wave * 16) * NTOK;
  const unsigned kLOff = (unsigned)(wave * 8 * CCH * 2);
  const unsigned vLOff = (unsigned)(wave * 16 * BC * 2);
  const unsigned kLds0 = (unsigned)(unsigned long long)&ktile[0][0][0] + kLOff;
  const unsigned kLds1 = (unsigned)(unsigned long long)&ktile[1][0][0] + kLOff;
  const unsigned vLds0 = (unsigned)(unsigned long long)&vtile[0][0][0] + vLOff;
  const unsigned vLds1 = (unsigned)(unsigned long long)&vtile[1][0][0] + vLOff;

  // Q fragments (A layout), 2 chunks of K-dim (c = 0..31, 32..63)
  HV aq[2];
  #pragma unroll
  for (int ct = 0; ct < 2; ++ct) {
    const _Float16* qp = qBase + (size_t)(i0 + m16) * CCH + ct * 32;
    aq[ct].h[0] = *(const v8h*)(qp + 16 * hi);
    aq[ct].h[1] = *(const v8h*)(qp + 16 * hi + 8);
  }

  v8f o_acc[4];
  #pragma unroll
  for (int ct = 0; ct < 4; ++ct)
    #pragma unroll
    for (int e = 0; e < 8; ++e) o_acc[ct][e] = 0.f;

  float mrow[8], lrow[8];
  #pragma unroll
  for (int r = 0; r < 8; ++r) { mrow[r] = -1e30f; lrow[r] = 0.f; }

  // Prologue: every wave DMAs its quarter of tile j0=0 into buffer 0.
  tdm_load_2d_f16(kBase + kGOff, kLds0, CCH, 8, CCH, NTOK, CCH);
  tdm_load_2d_f16(vBase + vGOff, vLds0, BC, 16, NTOK, CCH, NTOK);

  int buf = 0;
  for (int j0 = 0; j0 < NTOK; j0 += BC) {
    if ((j0 + BC) < NTOK) {   // uniform: issue next tile, then wait for cur
      tdm_load_2d_f16(kBase + kGOff + (size_t)(j0 + BC) * CCH,
                      buf ? kLds0 : kLds1, CCH, 8, CCH, NTOK, CCH);
      tdm_load_2d_f16(vBase + vGOff + (j0 + BC),
                      buf ? vLds0 : vLds1, BC, 16, NTOK, CCH, NTOK);
      __builtin_amdgcn_s_wait_tensorcnt(2);   // in-order: cur quarters landed
    } else {
      __builtin_amdgcn_s_wait_tensorcnt(0);
    }
    __syncthreads();   // publish all quarters to all waves

    // ----- load ALL B fragments for this tile up front (one DS clause) -----
    HV kf[2][2], vf[4];
    #pragma unroll
    for (int jt = 0; jt < 2; ++jt)
      #pragma unroll
      for (int ct = 0; ct < 2; ++ct) {
        const _Float16* kp = &ktile[buf][jt * 16 + m16][ct * 32 + hi * 16];
        kf[jt][ct].h[0] = *(const v8h*)(kp);
        kf[jt][ct].h[1] = *(const v8h*)(kp + 8);
      }
    #pragma unroll
    for (int ct = 0; ct < 4; ++ct) {
      const _Float16* vp = &vtile[buf][ct * 16 + m16][hi * 16];
      vf[ct].h[0] = *(const v8h*)(vp);
      vf[ct].h[1] = *(const v8h*)(vp + 8);
    }
    // Scheduling fence: keep the 16 ds_load_b128 batched ahead of the WMMA
    // chain (one staged s_wait_dscnt instead of a wait-0 per WMMA).
    asm volatile("" ::: "memory");

    // ----- S = Q^T K : 2 j-subtiles x 2 c-chunks -----
    v8f s[2];
    #pragma unroll
    for (int jt = 0; jt < 2; ++jt) {
      v8f acc;
      #pragma unroll
      for (int e = 0; e < 8; ++e) acc[e] = 0.f;
      #pragma unroll
      for (int ct = 0; ct < 2; ++ct)
        acc = __builtin_amdgcn_wmma_f32_16x16x32_f16(
            false, aq[ct].v, false, kf[jt][ct].v, (short)0, acc, false, false);
      s[jt] = acc;
    }

    // ----- online softmax (rows live in 16-lane halves) -----
    v8f mx;
    #pragma unroll
    for (int e = 0; e < 8; ++e) mx[e] = fmaxf(s[0][e], s[1][e]);
    #pragma unroll
    for (int off = 1; off <= 8; off <<= 1)
      #pragma unroll
      for (int e = 0; e < 8; ++e)
        mx[e] = fmaxf(mx[e], __shfl_xor(mx[e], off, 32));

    float scl[8];
    #pragma unroll
    for (int r = 0; r < 8; ++r) {
      float mnew = fmaxf(mrow[r], mx[r]);
      scl[r]  = __expf(mrow[r] - mnew);
      mrow[r] = mnew;
    }
    #pragma unroll
    for (int jt = 0; jt < 2; ++jt)
      #pragma unroll
      for (int r = 0; r < 8; ++r)
        s[jt][r] = __expf(s[jt][r] - mrow[r]);

    v8f ps;
    #pragma unroll
    for (int e = 0; e < 8; ++e) ps[e] = s[0][e] + s[1][e];
    #pragma unroll
    for (int off = 1; off <= 8; off <<= 1)
      #pragma unroll
      for (int e = 0; e < 8; ++e)
        ps[e] += __shfl_xor(ps[e], off, 32);

    #pragma unroll
    for (int r = 0; r < 8; ++r) lrow[r] = lrow[r] * scl[r] + ps[r];
    #pragma unroll
    for (int ct = 0; ct < 4; ++ct)
      #pragma unroll
      for (int r = 0; r < 8; ++r) o_acc[ct][r] *= scl[r];

    // ----- P: D layout -> A layout via wave-private LDS -----
    #pragma unroll
    for (int jt = 0; jt < 2; ++jt)
      #pragma unroll
      for (int r = 0; r < 8; ++r)
        pbuf[wave][r + 8 * hi][jt * 16 + m16] = (_Float16)s[jt][r];
    asm volatile("s_wait_dscnt 0" ::: "memory");   // DS in-order per wave
    HV ap;
    ap.h[0] = *(const v8h*)&pbuf[wave][m16][16 * hi];
    ap.h[1] = *(const v8h*)&pbuf[wave][m16][16 * hi + 8];

    // ----- O += P * V^T : 4 c-tiles, K = 32 -----
    #pragma unroll
    for (int ct = 0; ct < 4; ++ct)
      o_acc[ct] = __builtin_amdgcn_wmma_f32_16x16x32_f16(
          false, ap.v, false, vf[ct].v, (short)0, o_acc[ct], false, false);

    __syncthreads();   // all waves done with buf before TDM refills it
    buf ^= 1;
  }

  // ----- epilogue: out[b][c][i] = O / l -----
  #pragma unroll
  for (int ct = 0; ct < 4; ++ct) {
    #pragma unroll
    for (int r = 0; r < 8; ++r) {
      int c = ct * 16 + m16;
      int i = i0 + r + 8 * hi;
      out[(((size_t)b * CCH + c) << 12) + i] = o_acc[ct][r] / lrow[r];
    }
  }
}

// ---------------------------------------------------------------------------
extern "C" void kernel_launch(void* const* d_in, const int* in_sizes, int n_in,
                              void* d_out, int out_size, void* d_ws, size_t ws_size,
                              hipStream_t stream) {
  const float* x  = (const float*)d_in[0];
  const float* Wq = (const float*)d_in[1];
  const float* bq = (const float*)d_in[2];
  const float* Wk = (const float*)d_in[3];
  const float* bk = (const float*)d_in[4];
  const float* Wv = (const float*)d_in[5];
  const float* bv = (const float*)d_in[6];
  float* out = (float*)d_out;

  const size_t mat = (size_t)BATCH * NTOK * CCH;   // elements per f16 matrix
  _Float16* qT = (_Float16*)d_ws;
  _Float16* kT = qT + mat;
  _Float16* vC = kT + mat;

  proj_qkv_kernel<<<dim3(NTOK / 64, BATCH), 256, 0, stream>>>(
      x, Wq, bq, Wk, bk, Wv, bv, qT, kT, vC);
  flash_attn_kernel<<<dim3(NTOK / BR, BATCH), FLASH_WAVES * 32, 0, stream>>>(
      qT, kT, vC, out);
}
